// GraniteMoeMoE_36051955483080
// MI455X (gfx1250) — matrix operand order repaired and checked by reading, not compiled, not run on hardware
//
#include <hip/hip_runtime.h>
#include <hip/hip_bf16.h>
#include <math.h>

// ---------------------------------------------------------------------------
// GraniteMoE (top-2 of 8 experts, SwiGLU) for gfx1250.
// bf16 WMMA GEMMs with double-buffered GLOBAL_LOAD_ASYNC_TO_LDS staging.
// ---------------------------------------------------------------------------

typedef __bf16 bf16_t;
typedef __attribute__((ext_vector_type(16))) __bf16 v16bf;
typedef __attribute__((ext_vector_type(8)))  float  v8f;
typedef __attribute__((ext_vector_type(4)))  int    v4i;

constexpr int D_   = 1024;         // hidden
constexpr int H_   = 1024;         // intermediate
constexpr int E_   = 8;            // experts
constexpr int K_   = 2;            // top-k
constexpr int T_   = 2 * 1024;     // tokens
constexpr int P_   = T_ * K_;      // token-expert pairs (always exactly T*K)
constexpr int PAD_ = 128;          // tile-edge padding rows

constexpr int BM = 128, BN = 64;
constexpr int KT_ = 64;            // K elements staged per LDS buffer (2 WMMA steps)
constexpr int STR = 72;            // LDS row stride in bf16 (144B) -> conflict-friendly
constexpr int MT_MAX = T_ / BM;    // 16 worst-case M tiles per expert

#define AS1 __attribute__((address_space(1)))
#define AS3 __attribute__((address_space(3)))

// ---- async copy global -> LDS (16B per call), three-tier fallback ----------
#if __has_builtin(__builtin_amdgcn_global_load_async_to_lds_b128) && \
    __has_builtin(__builtin_amdgcn_s_wait_asynccnt)
#define ASYNC_MODE 2
#elif !defined(FORCE_SYNC_COPY)
#define ASYNC_MODE 1
#else
#define ASYNC_MODE 0
#endif

__device__ inline void cp16(const bf16_t* __restrict__ g, bf16_t* __restrict__ l) {
#if ASYNC_MODE == 2
  // prototype: (v4i __device__* src, v4i __shared__* dst, imm offset, imm cpol)
  AS1 v4i* gp = (AS1 v4i*)(unsigned long long)(size_t)g;   // global bit pattern
  AS3 v4i* lp = (AS3 v4i*)(unsigned)(size_t)l;             // LDS offset = addr[31:0]
  __builtin_amdgcn_global_load_async_to_lds_b128(gp, lp, 0, 0);
#elif ASYNC_MODE == 1
  asm volatile("global_load_async_to_lds_b128 %0, %1, off"
               :
               : "v"((unsigned)(unsigned long long)l), "v"(g)
               : "memory");
#else
  *(uint4*)l = *(const uint4*)g;
#endif
}

#if ASYNC_MODE == 2
#define ASYNC_WAIT(n) __builtin_amdgcn_s_wait_asynccnt(n)
#elif ASYNC_MODE == 1
#define ASYNC_WAIT(n) asm volatile("s_wait_asynccnt %0" ::"i"(n))
#else
#define ASYNC_WAIT(n)
#endif

union FragU { v16bf v; uint4 q[2]; };

__device__ inline bf16_t f2bf(float f) { return (bf16_t)f; }

// A operand (16x32 bf16, M=lane%16): lane holds K = {c..c+7, c+16..c+23}, c=lane<16?0:8
__device__ inline v16bf lds_a_frag(const bf16_t* __restrict__ s, int row0, int kl) {
  const int lane = threadIdx.x & 31;
  const int m    = lane & 15;
  const int c    = (lane < 16) ? 0 : 8;
  const bf16_t* p = s + (row0 + m) * STR + kl + c;
  FragU u;
  u.q[0] = *(const uint4*)(p);
  u.q[1] = *(const uint4*)(p + 16);
  return u.v;
}

// B operand (32x16 bf16, N=lane%16): lane holds K = {kk..kk+7, kk+16..kk+23}, kk=lane<16?0:8
__device__ inline v16bf lds_b_frag(const bf16_t* __restrict__ s, int row0, int kl) {
  const int lane = threadIdx.x & 31;
  const int n    = lane & 15;
  const int kk   = kl + ((lane < 16) ? 0 : 8);
  const bf16_t* p = s + (row0 + n) * STR + kk;
  FragU u;
  u.q[0] = *(const uint4*)(p);
  u.q[1] = *(const uint4*)(p + 16);
  return u.v;
}

__device__ inline v8f wmma_bf16(v16bf a, v16bf b, v8f c) {
  return __builtin_amdgcn_wmma_f32_16x16x32_bf16(false, a, false, b, (short)0, c,
                                                 false, false);
}

// ---------------------------------------------------------------------------
// 0) zero the tiny counters
__global__ void moe_init(int* __restrict__ counts, int* __restrict__ fill) {
  int i = threadIdx.x;
  if (i < E_) { counts[i] = 0; fill[i] = 0; }
}

// 1) fp32 -> bf16 cast (vectorized 4-wide)
struct bf4 { bf16_t a, b, c, d; };
__global__ __launch_bounds__(256)
void cast_f32_bf16(const float* __restrict__ src, bf16_t* __restrict__ dst, int n) {
  int i = (blockIdx.x * blockDim.x + threadIdx.x) * 4;
  if (i < n) {
    float4 f = *(const float4*)(src + i);
    bf4 o = { f2bf(f.x), f2bf(f.y), f2bf(f.z), f2bf(f.w) };
    *(bf4*)(dst + i) = o;
  }
}

// 2) router: one block per token, one wave per expert (E_ == 8 waves)
__global__ __launch_bounds__(256)
void moe_router(const float* __restrict__ x, const float* __restrict__ wr,
                int* __restrict__ topk_idx, float* __restrict__ topk_gate,
                int* __restrict__ counts) {
  const int t    = blockIdx.x;
  const int wave = threadIdx.x >> 5;
  const int lane = threadIdx.x & 31;
  const float* xt = x  + (size_t)t    * D_;
  const float* we = wr + (size_t)wave * D_;
  float acc = 0.f;
  for (int d = lane; d < D_; d += 32) acc += xt[d] * we[d];
  for (int off = 16; off > 0; off >>= 1) acc += __shfl_xor(acc, off, 32);
  __shared__ float sl[E_];
  if (lane == 0) sl[wave] = acc;
  __syncthreads();
  if (threadIdx.x == 0) {
    int i0 = 0; float v0 = sl[0];
    for (int e = 1; e < E_; e++) if (sl[e] > v0) { v0 = sl[e]; i0 = e; }
    int i1 = (i0 == 0) ? 1 : 0; float v1 = sl[i1];
    for (int e = 0; e < E_; e++) if (e != i0 && sl[e] > v1) { v1 = sl[e]; i1 = e; }
    float ee  = __expf(v1 - v0);          // softmax over the two top logits
    float inv = 1.f / (1.f + ee);
    topk_idx[2 * t]     = i0;  topk_gate[2 * t]     = inv;
    topk_idx[2 * t + 1] = i1;  topk_gate[2 * t + 1] = ee * inv;
    atomicAdd(&counts[i0], 1);
    atomicAdd(&counts[i1], 1);
  }
}

// 3) exclusive prefix over 8 expert counts
__global__ void moe_scan(const int* __restrict__ counts, int* __restrict__ base_) {
  if (threadIdx.x == 0 && blockIdx.x == 0) {
    int a = 0;
    for (int e = 0; e < E_; e++) { base_[e] = a; a += counts[e]; }
  }
}

// 4) scatter (token,k) pairs into expert-grouped slots
__global__ __launch_bounds__(256)
void moe_scatter(const int* __restrict__ topk_idx, const int* __restrict__ base_,
                 int* __restrict__ fill, int* __restrict__ tok_of,
                 int* __restrict__ slot_of) {
  int t = blockIdx.x * blockDim.x + threadIdx.x;
  if (t < T_) {
    for (int k = 0; k < K_; k++) {
      int e = topk_idx[2 * t + k];
      int s = base_[e] + atomicAdd(&fill[e], 1);
      tok_of[s] = t;
      slot_of[2 * t + k] = s;
    }
  }
}

// 5) gather token rows into slot order, fp32 -> bf16
__global__ __launch_bounds__(256)
void moe_gather(const float* __restrict__ x, const int* __restrict__ tok_of,
                bf16_t* __restrict__ xg) {
  const int s = blockIdx.x;                       // 0..P_-1
  const int t = tok_of[s];
  const float* src = x  + (size_t)t * D_;
  bf16_t*      dst = xg + (size_t)s * D_;
  int d = threadIdx.x * 4;                        // 256 threads * 4 == D_
  float4 f = *(const float4*)(src + d);
  bf4 o = { f2bf(f.x), f2bf(f.y), f2bf(f.z), f2bf(f.w) };
  *(bf4*)(dst + d) = o;
}

// ---------------------------------------------------------------------------
// 6) GEMM1: g = silu(xg @ Wa^T) * (xg @ Wb^T) per expert.
//    Block tile 128(M) x 64(N); 8 waves in 4x2; double-buffered async LDS.
//    Bs rows 0..63 = a-half, 64..127 = b-half. 8 async b128 per thread/buffer.
__global__ __launch_bounds__(256)
void moe_up_swiglu(const bf16_t* __restrict__ xg, const bf16_t* __restrict__ w_in,
                   const int* __restrict__ base_, const int* __restrict__ counts,
                   bf16_t* __restrict__ g) {
  const int e   = blockIdx.z;
  const int cnt = counts[e];
  const int m0  = blockIdx.y * BM;
  if (m0 >= cnt) return;
  const int n0  = blockIdx.x * BN;
  const int b0  = base_[e];
  const int wave = threadIdx.x >> 5;
  const int wm = wave >> 1, wn = wave & 1;

  __shared__ __align__(16) bf16_t As[2][BM * STR];
  __shared__ __align__(16) bf16_t Bs[2][BM * STR];

  const bf16_t* Ag = xg + (size_t)(b0 + m0) * D_;
  const bf16_t* Wa = w_in + ((size_t)e * 2 * H_ + n0) * D_;
  const bf16_t* Wb = Wa + (size_t)H_ * D_;

  auto stage = [&](int buf, int k) {
    // A tile: 128 rows x 64 K-elems = 1024 16B chunks -> 4 per thread
    for (int j = 0; j < 4; ++j) {
      int c = threadIdx.x + j * 256;
      int row = c >> 3, col = (c & 7) * 8;
      cp16(Ag + (size_t)row * D_ + k + col, &As[buf][row * STR + col]);
    }
    // B tile: 64 a-rows + 64 b-rows -> 4 per thread
    for (int j = 0; j < 4; ++j) {
      int c = threadIdx.x + j * 256;
      int row = c >> 3, col = (c & 7) * 8;
      const bf16_t* src = (row < 64) ? (Wa + (size_t)row * D_ + k + col)
                                     : (Wb + (size_t)(row - 64) * D_ + k + col);
      cp16(src, &Bs[buf][row * STR + col]);
    }
  };

  const v8f zero = {0.f, 0.f, 0.f, 0.f, 0.f, 0.f, 0.f, 0.f};
  v8f accA[2][2], accB[2][2];
  for (int i = 0; i < 2; i++)
    for (int j = 0; j < 2; j++) { accA[i][j] = zero; accB[i][j] = zero; }

  constexpr int KTILES = D_ / KT_;      // 16
  stage(0, 0);
  int buf = 0;
  for (int kt = 0; kt < KTILES; ++kt) {
    if (kt + 1 < KTILES) {
      stage(buf ^ 1, (kt + 1) * KT_);
      ASYNC_WAIT(8);                    // previous buffer's 8 copies complete
    } else {
      ASYNC_WAIT(0);
    }
    __syncthreads();
#pragma unroll
    for (int ks = 0; ks < 2; ++ks) {
      const int kl = ks * 32;
      v16bf a0  = lds_a_frag(As[buf], wm * 32,      kl);
      v16bf a1  = lds_a_frag(As[buf], wm * 32 + 16, kl);
      v16bf ba0 = lds_b_frag(Bs[buf], wn * 32,      kl);
      v16bf ba1 = lds_b_frag(Bs[buf], wn * 32 + 16, kl);
      v16bf bb0 = lds_b_frag(Bs[buf], 64 + wn * 32,      kl);
      v16bf bb1 = lds_b_frag(Bs[buf], 64 + wn * 32 + 16, kl);
      accA[0][0] = wmma_bf16(a0, ba0, accA[0][0]);
      accA[0][1] = wmma_bf16(a0, ba1, accA[0][1]);
      accA[1][0] = wmma_bf16(a1, ba0, accA[1][0]);
      accA[1][1] = wmma_bf16(a1, ba1, accA[1][1]);
      accB[0][0] = wmma_bf16(a0, bb0, accB[0][0]);
      accB[0][1] = wmma_bf16(a0, bb1, accB[0][1]);
      accB[1][0] = wmma_bf16(a1, bb0, accB[1][0]);
      accB[1][1] = wmma_bf16(a1, bb1, accB[1][1]);
    }
    __syncthreads();                    // buffer reuse fence
    buf ^= 1;
  }

  const int lane  = threadIdx.x & 31;
  const int colL  = lane & 15;
  const int rbase = (lane < 16) ? 0 : 8;
  for (int mi = 0; mi < 2; mi++)
    for (int ni = 0; ni < 2; ni++)
      for (int r = 0; r < 8; r++) {
        int mrow = m0 + wm * 32 + mi * 16 + rbase + r;
        if (mrow < cnt) {
          float av = accA[mi][ni][r];
          float bv = accB[mi][ni][r];
          float gv = (av / (1.f + __expf(-av))) * bv;   // silu(a) * b
          int col = n0 + wn * 32 + ni * 16 + colL;
          g[(size_t)(b0 + mrow) * H_ + col] = f2bf(gv);
        }
      }
}

// 7) GEMM2: po = g @ w_out[e]^T (N over D). Same staging, 6 copies/thread/buffer.
__global__ __launch_bounds__(256)
void moe_down(const bf16_t* __restrict__ g, const bf16_t* __restrict__ w_out,
              const int* __restrict__ base_, const int* __restrict__ counts,
              float* __restrict__ po) {
  const int e   = blockIdx.z;
  const int cnt = counts[e];
  const int m0  = blockIdx.y * BM;
  if (m0 >= cnt) return;
  const int n0  = blockIdx.x * BN;
  const int b0  = base_[e];
  const int wave = threadIdx.x >> 5;
  const int wm = wave >> 1, wn = wave & 1;

  __shared__ __align__(16) bf16_t As[2][BM * STR];
  __shared__ __align__(16) bf16_t Bs[2][BN * STR];

  const bf16_t* Ag = g + (size_t)(b0 + m0) * H_;
  const bf16_t* Wg = w_out + ((size_t)e * D_ + n0) * H_;

  auto stage = [&](int buf, int k) {
    for (int j = 0; j < 4; ++j) {
      int c = threadIdx.x + j * 256;
      int row = c >> 3, col = (c & 7) * 8;
      cp16(Ag + (size_t)row * H_ + k + col, &As[buf][row * STR + col]);
    }
    for (int j = 0; j < 2; ++j) {
      int c = threadIdx.x + j * 256;
      int row = c >> 3, col = (c & 7) * 8;
      cp16(Wg + (size_t)row * H_ + k + col, &Bs[buf][row * STR + col]);
    }
  };

  const v8f zero = {0.f, 0.f, 0.f, 0.f, 0.f, 0.f, 0.f, 0.f};
  v8f acc[2][2];
  for (int i = 0; i < 2; i++)
    for (int j = 0; j < 2; j++) acc[i][j] = zero;

  constexpr int KTILES = H_ / KT_;      // 16
  stage(0, 0);
  int buf = 0;
  for (int kt = 0; kt < KTILES; ++kt) {
    if (kt + 1 < KTILES) {
      stage(buf ^ 1, (kt + 1) * KT_);
      ASYNC_WAIT(6);
    } else {
      ASYNC_WAIT(0);
    }
    __syncthreads();
#pragma unroll
    for (int ks = 0; ks < 2; ++ks) {
      const int kl = ks * 32;
      v16bf a0 = lds_a_frag(As[buf], wm * 32,      kl);
      v16bf a1 = lds_a_frag(As[buf], wm * 32 + 16, kl);
      v16bf b0f = lds_b_frag(Bs[buf], wn * 32,      kl);
      v16bf b1f = lds_b_frag(Bs[buf], wn * 32 + 16, kl);
      acc[0][0] = wmma_bf16(a0, b0f, acc[0][0]);
      acc[0][1] = wmma_bf16(a0, b1f, acc[0][1]);
      acc[1][0] = wmma_bf16(a1, b0f, acc[1][0]);
      acc[1][1] = wmma_bf16(a1, b1f, acc[1][1]);
    }
    __syncthreads();
    buf ^= 1;
  }

  const int lane  = threadIdx.x & 31;
  const int colL  = lane & 15;
  const int rbase = (lane < 16) ? 0 : 8;
  for (int mi = 0; mi < 2; mi++)
    for (int ni = 0; ni < 2; ni++)
      for (int r = 0; r < 8; r++) {
        int mrow = m0 + wm * 32 + mi * 16 + rbase + r;
        if (mrow < cnt) {
          int col = n0 + wn * 32 + ni * 16 + colL;
          po[(size_t)(b0 + mrow) * D_ + col] = acc[mi][ni][r];
        }
      }
}

// 8) final combine: out[t] = gate0*po[slot0] + gate1*po[slot1]
__global__ __launch_bounds__(256)
void moe_combine(const float* __restrict__ po, const int* __restrict__ slot_of,
                 const float* __restrict__ gates, float* __restrict__ out) {
  const int t  = blockIdx.x;
  const int s0 = slot_of[2 * t], s1 = slot_of[2 * t + 1];
  const float g0 = gates[2 * t], g1 = gates[2 * t + 1];
  int d = threadIdx.x * 4;
  float4 p0 = *(const float4*)(po + (size_t)s0 * D_ + d);
  float4 p1 = *(const float4*)(po + (size_t)s1 * D_ + d);
  float4 o;
  o.x = g0 * p0.x + g1 * p1.x;
  o.y = g0 * p0.y + g1 * p1.y;
  o.z = g0 * p0.z + g1 * p1.z;
  o.w = g0 * p0.w + g1 * p1.w;
  *(float4*)(out + (size_t)t * D_ + d) = o;
}

// ---------------------------------------------------------------------------
extern "C" void kernel_launch(void* const* d_in, const int* in_sizes, int n_in,
                              void* d_out, int out_size, void* d_ws, size_t ws_size,
                              hipStream_t stream) {
  const float* x        = (const float*)d_in[0];   // [B,S,D]
  const float* w_router = (const float*)d_in[1];   // [E,D]
  const float* w_in     = (const float*)d_in[2];   // [E,2H,D]
  const float* w_out    = (const float*)d_in[3];   // [E,D,H]
  float* out            = (float*)d_out;           // [B,S,D]

  // ---- workspace carve-up (256B aligned regions) ----
  char* w = (char*)d_ws;
  size_t off = 0;
  auto carve = [&](size_t bytes) -> void* {
    off = (off + 255) & ~(size_t)255;
    void* p = w + off;
    off += bytes;
    return p;
  };
  bf16_t* w_in_bf  = (bf16_t*)carve((size_t)E_ * 2 * H_ * D_ * 2);
  bf16_t* w_out_bf = (bf16_t*)carve((size_t)E_ * D_ * H_ * 2);
  bf16_t* xg       = (bf16_t*)carve((size_t)(P_ + PAD_) * D_ * 2);
  bf16_t* gbuf     = (bf16_t*)carve((size_t)(P_ + PAD_) * H_ * 2);
  float*  po       = (float*)carve((size_t)(P_ + PAD_) * D_ * 4);
  int*    topk_idx = (int*)carve((size_t)P_ * 4);
  float*  topk_gt  = (float*)carve((size_t)P_ * 4);
  int*    slot_of  = (int*)carve((size_t)P_ * 4);
  int*    tok_of   = (int*)carve((size_t)(P_ + PAD_) * 4);
  int*    counts   = (int*)carve((size_t)E_ * 4);
  int*    fill     = (int*)carve((size_t)E_ * 4);
  int*    base_    = (int*)carve((size_t)E_ * 4);

  // ---- pipeline ----
  moe_init<<<1, 64, 0, stream>>>(counts, fill);

  {
    int n1 = E_ * 2 * H_ * D_;
    cast_f32_bf16<<<n1 / (256 * 4), 256, 0, stream>>>(w_in, w_in_bf, n1);
    int n2 = E_ * D_ * H_;
    cast_f32_bf16<<<n2 / (256 * 4), 256, 0, stream>>>(w_out, w_out_bf, n2);
  }

  moe_router<<<T_, 256, 0, stream>>>(x, w_router, topk_idx, topk_gt, counts);
  moe_scan<<<1, 1, 0, stream>>>(counts, base_);
  moe_scatter<<<T_ / 256, 256, 0, stream>>>(topk_idx, base_, fill, tok_of, slot_of);
  moe_gather<<<P_, 256, 0, stream>>>(x, tok_of, xg);

  {
    dim3 grid1(H_ / BN, MT_MAX, E_);          // 16 x 16 x 8
    moe_up_swiglu<<<grid1, 256, 0, stream>>>(xg, w_in_bf, base_, counts, gbuf);
    dim3 grid2(D_ / BN, MT_MAX, E_);
    moe_down<<<grid2, 256, 0, stream>>>(gbuf, w_out_bf, base_, counts, po);
  }

  moe_combine<<<T_, 256, 0, stream>>>(po, slot_of, topk_gt, out);
}
